// Model_19774029431282
// MI455X (gfx1250) — compile-verified, matrix operand order checked by reading, not run on hardware
//
#include <hip/hip_runtime.h>
#include <hip/hip_bf16.h>

// ---------------------------------------------------------------------------
// BiLSTM-CRF forward (score + Viterbi path) for gfx1250 (MI455X, wave32).
// Pipeline:
//   1) embed_gather      : x = emb[sentences] -> bf16          [B,S,E]
//   2) f32_to_bf16       : Wih/Whh weights -> bf16; w_out padded 12->16 rows
//   3) gates_gemm (WMMA) : G = x @ Wih^T + (bih+bhh) -> bf16   [B,S,4H]  (x2 dir)
//   4) lstm_rec   (WMMA) : persistent 1-WG-per-direction recurrence,
//                          h double-buffered in LDS, c in VGPRs
//   5) feats_gemm (WMMA) : feats = [hf|hb] @ w_out^T + b_out   [B,S,T]
//   6) viterbi_kernel    : CRF DP + backtrace, 1 wave per batch element
// ---------------------------------------------------------------------------

#define B_ 64
#define S_ 512
#define E_ 256
#define H_ 256
#define G4H 1024
#define T_ 12
#define TPAD 16
#define START_ 10
#define STOP_ 11
#define NEG_ -10000.0f

typedef __attribute__((ext_vector_type(16))) __bf16 v16bf;
typedef __attribute__((ext_vector_type(8)))  float  v8f;

union Frag16 { unsigned u[8]; v16bf v; };

static __device__ __forceinline__ unsigned short f2bf(float f) {
  unsigned u = __float_as_uint(f);
  u += 0x7FFFu + ((u >> 16) & 1u);          // round-to-nearest-even
  return (unsigned short)(u >> 16);
}
static __device__ __forceinline__ float bf2f(unsigned short h) {
  return __uint_as_float(((unsigned)h) << 16);
}

// --------------------------------------------------------------------------
__global__ void f32_to_bf16_kernel(const float* __restrict__ src,
                                   unsigned short* __restrict__ dst, int n) {
  int i = blockIdx.x * 256 + threadIdx.x;
  if (i < n) dst[i] = f2bf(src[i]);
}

// w_out [12,512] f32 -> wpad [16,512] bf16, rows 12..15 zeroed
__global__ void pad_wout_kernel(const float* __restrict__ w_out,
                                unsigned short* __restrict__ wpad) {
  int i = blockIdx.x * 256 + threadIdx.x;   // 0 .. 16*512-1
  int row = i >> 9;
  wpad[i] = (row < T_) ? f2bf(w_out[i]) : (unsigned short)0;
}

// x[b,s,:] = bf16(emb[sentences[b,s],:])
__global__ void embed_gather(const int* __restrict__ sent,
                             const float* __restrict__ emb,
                             unsigned short* __restrict__ xbf) {
  int bs = blockIdx.x;                 // 0 .. B*S-1
  int e  = threadIdx.x;                // 0 .. 255
  int tok = sent[bs];
  xbf[bs * E_ + e] = f2bf(emb[tok * E_ + e]);
}

// --------------------------------------------------------------------------
// G[M=32768, N=1024] = X[M,256](bf16) @ W[1024,256]^T(bf16) + (bih+bhh)
// One 16x16 output tile per wave; 8 waves per block; K-loop of 8 WMMAs.
__global__ __launch_bounds__(256) void gates_gemm(
    const unsigned short* __restrict__ X,
    const unsigned short* __restrict__ W,
    const float* __restrict__ bih, const float* __restrict__ bhh,
    unsigned short* __restrict__ G) {
  const int lane = threadIdx.x & 31;
  const int wave = threadIdx.x >> 5;
  const int half = lane >> 4;
  const int lm   = lane & 15;
  const int n0 = blockIdx.x * 16;                  // gridDim.x = 64
  const int m0 = (blockIdx.y * 8 + wave) * 16;     // gridDim.y = 256

  v8f acc = {0.f, 0.f, 0.f, 0.f, 0.f, 0.f, 0.f, 0.f};
  const int rowA = (m0 + lm) * E_;
  const int rowB = (n0 + lm) * E_;
#pragma unroll 1
  for (int k0 = 0; k0 < E_; k0 += 32) {
    Frag16 a, b;
#pragma unroll
    for (int v = 0; v < 8; ++v) {
      int k = k0 + ((v & 4) << 2) + (half << 3) + ((v & 3) << 1);
      a.u[v] = *(const unsigned*)(X + rowA + k);
      b.u[v] = *(const unsigned*)(W + rowB + k);
    }
    acc = __builtin_amdgcn_wmma_f32_16x16x32_bf16(
        false, a.v, false, b.v, (short)0, acc, false, false);
  }
  const int n = n0 + lm;
  const float bias = bih[n] + bhh[n];
#pragma unroll
  for (int v = 0; v < 8; ++v) {
    int m = m0 + v + (half << 3);
    G[m * G4H + n] = f2bf(acc[v] + bias);
  }
}

// --------------------------------------------------------------------------
// Persistent recurrence: blockIdx.x = direction. 256 threads = 8 waves.
// Wave w owns h-columns [w*32, w*32+32); per m-chunk it computes the 8
// gate tiles (4 gates x 2 n-subtiles) whose elementwise combination updates
// exactly its private c fragment (kept in VGPRs).
__global__ __launch_bounds__(256) void lstm_rec(
    const unsigned short* __restrict__ Gf, const unsigned short* __restrict__ Gb,
    const unsigned short* __restrict__ whhf, const unsigned short* __restrict__ whhb,
    const float* __restrict__ h0, const float* __restrict__ c0,
    unsigned short* __restrict__ hfs, unsigned short* __restrict__ hbs) {
  extern __shared__ unsigned short hbuf[];   // 2 * 64 * 256 bf16 = 64 KB
  const int dir = blockIdx.x;
  const unsigned short* G   = dir ? Gb  : Gf;
  const unsigned short* whh = dir ? whhb : whhf;
  unsigned short* hs        = dir ? hbs : hfs;

  const int lane = threadIdx.x & 31;
  const int wave = threadIdx.x >> 5;
  const int half = lane >> 4;
  const int lm   = lane & 15;

  // h(0) -> LDS buffer 0
  for (int i = threadIdx.x; i < B_ * H_; i += 256)
    hbuf[i] = f2bf(h0[dir * B_ * H_ + i]);

  // c(0) -> registers, matching the C/D fragment layout
  float c[4][2][8];
#pragma unroll
  for (int mi = 0; mi < 4; ++mi)
#pragma unroll
    for (int t2 = 0; t2 < 2; ++t2)
#pragma unroll
      for (int v = 0; v < 8; ++v) {
        int m = mi * 16 + v + (half << 3);
        int n = wave * 32 + t2 * 16 + lm;
        c[mi][t2][v] = c0[dir * B_ * H_ + m * H_ + n];
      }
  __syncthreads();

#pragma unroll 1
  for (int t = 0; t < S_; ++t) {
    const unsigned short* hcur = hbuf + (t & 1) * B_ * H_;
    unsigned short*       hnxt = hbuf + ((t + 1) & 1) * B_ * H_;
    const int ts = dir ? (S_ - 1 - t) : t;

#pragma unroll 1
    for (int mi = 0; mi < 4; ++mi) {
      v8f acc[4][2];
#pragma unroll
      for (int g = 0; g < 4; ++g)
#pragma unroll
        for (int t2 = 0; t2 < 2; ++t2)
          acc[g][t2] = (v8f){0.f, 0.f, 0.f, 0.f, 0.f, 0.f, 0.f, 0.f};

      const int rowA = (mi * 16 + lm) * H_;
#pragma unroll 1
      for (int k0 = 0; k0 < H_; k0 += 32) {
        Frag16 a;
#pragma unroll
        for (int v = 0; v < 8; ++v) {
          int k = k0 + ((v & 4) << 2) + (half << 3) + ((v & 3) << 1);
          a.u[v] = *(const unsigned*)(hcur + rowA + k);
        }
#pragma unroll
        for (int g = 0; g < 4; ++g)
#pragma unroll
          for (int t2 = 0; t2 < 2; ++t2) {
            const int n0 = g * H_ + wave * 32 + t2 * 16;
            Frag16 b;
#pragma unroll
            for (int v = 0; v < 8; ++v) {
              int k = k0 + ((v & 4) << 2) + (half << 3) + ((v & 3) << 1);
              b.u[v] = *(const unsigned*)(whh + (n0 + lm) * H_ + k);
            }
            acc[g][t2] = __builtin_amdgcn_wmma_f32_16x16x32_bf16(
                false, a.v, false, b.v, (short)0, acc[g][t2], false, false);
          }
      }

      // add precomputed input gates, apply activations, update c, emit h
#pragma unroll
      for (int t2 = 0; t2 < 2; ++t2) {
#pragma unroll
        for (int v = 0; v < 8; ++v) {
          int m  = mi * 16 + v + (half << 3);
          int nh = wave * 32 + t2 * 16 + lm;
          int gbase = (m * S_ + ts) * G4H;
          float gi = acc[0][t2][v] + bf2f(G[gbase + 0 * H_ + nh]);
          float gf = acc[1][t2][v] + bf2f(G[gbase + 1 * H_ + nh]);
          float gg = acc[2][t2][v] + bf2f(G[gbase + 2 * H_ + nh]);
          float go = acc[3][t2][v] + bf2f(G[gbase + 3 * H_ + nh]);
          float ii = 1.f / (1.f + __expf(-gi));
          float ff = 1.f / (1.f + __expf(-gf));
          float tg = tanhf(gg);
          float oo = 1.f / (1.f + __expf(-go));
          float cn = ff * c[mi][t2][v] + ii * tg;
          c[mi][t2][v] = cn;
          unsigned short hb16 = f2bf(oo * tanhf(cn));
          hnxt[m * H_ + nh] = hb16;                    // next-step A operand
          hs[(m * S_ + ts) * H_ + nh] = hb16;          // h-sequence output
        }
      }
    }
    __syncthreads();   // publish hnxt before anyone reads it at t+1
  }
}

// --------------------------------------------------------------------------
// feats[bs, 0..11] = [hf[bs,:] | hb[bs,:]] @ wpad^T + b_out  via WMMA.
// M = B*S = 32768, N = 16 (T padded), K = 512. One M-tile per wave.
__global__ __launch_bounds__(256) void feats_gemm(
    const unsigned short* __restrict__ hf,
    const unsigned short* __restrict__ hb,
    const unsigned short* __restrict__ wpad,   // [16, 512] bf16
    const float* __restrict__ b_out,
    float* __restrict__ feats) {
  const int lane = threadIdx.x & 31;
  const int wave = threadIdx.x >> 5;
  const int half = lane >> 4;
  const int lm   = lane & 15;
  const int m0 = (blockIdx.x * 8 + wave) * 16;   // gridDim.x = 256

  v8f acc = {0.f, 0.f, 0.f, 0.f, 0.f, 0.f, 0.f, 0.f};
  const int rowA = (m0 + lm) * H_;               // row offset inside hf or hb
  const int rowB = lm * (2 * H_);                // row lm of wpad
#pragma unroll 1
  for (int k0 = 0; k0 < 2 * H_; k0 += 32) {
    // K chunks never straddle the hf/hb boundary (256 % 32 == 0)
    const unsigned short* baseA =
        (k0 < H_) ? (hf + rowA) : (hb + rowA - H_);
    Frag16 a, b;
#pragma unroll
    for (int v = 0; v < 8; ++v) {
      int k = k0 + ((v & 4) << 2) + (half << 3) + ((v & 3) << 1);
      a.u[v] = *(const unsigned*)(baseA + k);
      b.u[v] = *(const unsigned*)(wpad + rowB + k);
    }
    acc = __builtin_amdgcn_wmma_f32_16x16x32_bf16(
        false, a.v, false, b.v, (short)0, acc, false, false);
  }
  if (lm < T_) {                                 // drop the 4 pad columns
    const float bias = b_out[lm];
#pragma unroll
    for (int v = 0; v < 8; ++v) {
      int m = m0 + v + (half << 3);
      feats[m * T_ + lm] = acc[v] + bias;
    }
  }
}

// --------------------------------------------------------------------------
// CRF Viterbi: one wave per batch element; backpointers in LDS.
__global__ __launch_bounds__(32) void viterbi_kernel(
    const float* __restrict__ feats, const float* __restrict__ trans,
    float* __restrict__ out) {
  __shared__ float fv[T_];
  __shared__ float tr[T_ * T_];
  __shared__ unsigned char bp[S_][T_];
  const int b = blockIdx.x;
  const int j = threadIdx.x;

  for (int i = j; i < T_ * T_; i += 32) tr[i] = trans[i];
  if (j < T_) fv[j] = (j == START_) ? 0.f : NEG_;
  __syncthreads();

  for (int t = 0; t < S_; ++t) {
    float best = 0.f; int arg = 0;
    if (j < T_) {
      best = fv[0] + tr[j * T_ + 0];
      for (int k = 1; k < T_; ++k) {
        float v = fv[k] + tr[j * T_ + k];
        if (v > best) { best = v; arg = k; }   // first-index tie-break
      }
    }
    __syncthreads();
    if (j < T_) {
      fv[j] = best + feats[(b * S_ + t) * T_ + j];
      bp[t][j] = (unsigned char)arg;
    }
    __syncthreads();
  }

  if (j == 0) {
    float bestv = fv[0] + tr[STOP_ * T_ + 0]; int bi = 0;
    for (int k = 1; k < T_; ++k) {
      float v = fv[k] + tr[STOP_ * T_ + k];
      if (v > bestv) { bestv = v; bi = k; }
    }
    out[b] = bestv;                       // score
    int cur = bi;
    for (int t = S_ - 1; t >= 0; --t) {   // path[t] = cur; cur = bp[t][cur]
      out[B_ + b * S_ + t] = (float)cur;
      cur = bp[t][cur];
    }
  }
}

// --------------------------------------------------------------------------
extern "C" void kernel_launch(void* const* d_in, const int* in_sizes, int n_in,
                              void* d_out, int out_size, void* d_ws, size_t ws_size,
                              hipStream_t stream) {
  (void)in_sizes; (void)n_in; (void)out_size; (void)ws_size;
  const int*   sent  = (const int*)  d_in[0];
  const float* emb   = (const float*)d_in[1];
  const float* wih_f = (const float*)d_in[2];
  const float* whh_f = (const float*)d_in[3];
  const float* bih_f = (const float*)d_in[4];
  const float* bhh_f = (const float*)d_in[5];
  const float* wih_b = (const float*)d_in[6];
  const float* whh_b = (const float*)d_in[7];
  const float* bih_b = (const float*)d_in[8];
  const float* bhh_b = (const float*)d_in[9];
  const float* w_out = (const float*)d_in[10];
  const float* b_out = (const float*)d_in[11];
  const float* trans = (const float*)d_in[12];
  const float* h0    = (const float*)d_in[13];
  const float* c0    = (const float*)d_in[14];

  char* ws = (char*)d_ws;
  size_t off = 0;
  auto alloc = [&](size_t bytes) -> void* {
    void* p = ws + off;
    off += (bytes + 255) & ~(size_t)255;
    return p;
  };
  unsigned short* xbf     = (unsigned short*)alloc((size_t)B_ * S_ * E_ * 2);
  unsigned short* wihf_bf = (unsigned short*)alloc((size_t)G4H * E_ * 2);
  unsigned short* whhf_bf = (unsigned short*)alloc((size_t)G4H * H_ * 2);
  unsigned short* wihb_bf = (unsigned short*)alloc((size_t)G4H * E_ * 2);
  unsigned short* whhb_bf = (unsigned short*)alloc((size_t)G4H * H_ * 2);
  unsigned short* wpad    = (unsigned short*)alloc((size_t)TPAD * 2 * H_ * 2);
  unsigned short* Gf      = (unsigned short*)alloc((size_t)B_ * S_ * G4H * 2);
  unsigned short* Gb      = (unsigned short*)alloc((size_t)B_ * S_ * G4H * 2);
  unsigned short* hfs     = (unsigned short*)alloc((size_t)B_ * S_ * H_ * 2);
  unsigned short* hbs     = (unsigned short*)alloc((size_t)B_ * S_ * H_ * 2);
  float*          feats   = (float*)         alloc((size_t)B_ * S_ * T_ * 4);

  const int wn = G4H * E_;                  // 262144
  f32_to_bf16_kernel<<<(wn + 255) / 256, 256, 0, stream>>>(wih_f, wihf_bf, wn);
  f32_to_bf16_kernel<<<(wn + 255) / 256, 256, 0, stream>>>(whh_f, whhf_bf, wn);
  f32_to_bf16_kernel<<<(wn + 255) / 256, 256, 0, stream>>>(wih_b, wihb_bf, wn);
  f32_to_bf16_kernel<<<(wn + 255) / 256, 256, 0, stream>>>(whh_b, whhb_bf, wn);
  pad_wout_kernel<<<(TPAD * 2 * H_) / 256, 256, 0, stream>>>(w_out, wpad);

  embed_gather<<<B_ * S_, E_, 0, stream>>>(sent, emb, xbf);

  dim3 gg(G4H / 16, (B_ * S_) / (8 * 16));  // (64, 256)
  gates_gemm<<<gg, 256, 0, stream>>>(xbf, wihf_bf, bih_f, bhh_f, Gf);
  gates_gemm<<<gg, 256, 0, stream>>>(xbf, wihb_bf, bih_b, bhh_b, Gb);

  static int lds_attr_set = 0;  // idempotent, not a stream op (capture-safe)
  if (!lds_attr_set) {
    hipFuncSetAttribute((const void*)lstm_rec,
                        hipFuncAttributeMaxDynamicSharedMemorySize, 2 * B_ * H_ * 2);
    lds_attr_set = 1;
  }
  lstm_rec<<<2, 256, 2 * B_ * H_ * 2, stream>>>(Gf, Gb, whhf_bf, whhb_bf,
                                                h0, c0, hfs, hbs);

  feats_gemm<<<(B_ * S_) / (8 * 16), 256, 0, stream>>>(hfs, hbs, wpad, b_out, feats);

  viterbi_kernel<<<B_, 32, 0, stream>>>(feats, trans, (float*)d_out);
}
